// MLPSemanticLayer_17274358464774
// MI455X (gfx1250) — matrix operand order repaired and checked by reading, not compiled, not run on hardware
//
#include <hip/hip_runtime.h>

typedef __attribute__((ext_vector_type(16))) __bf16 v16bf;
typedef __attribute__((ext_vector_type(2)))  __bf16 v2bf;
typedef __attribute__((ext_vector_type(8)))  float  v8f;

#define F_IN   128
#define S_DIM  64
#define F_OUT  256
#define KROWS  16
#define WAVES  8
#define BN_PER_BLOCK 16
// padded row strides (32-bit units holding bf16 pairs)
#define W1_STRIDE 100   // 192 bf16 + pad
#define W2_STRIDE 36    // 64 bf16 + pad
#define HS_STRIDE 68    // 64 fp32 + pad (floats)

union Frag { unsigned u[8]; v16bf v; };

__device__ __forceinline__ unsigned pk_bf16(float a, float b) {
#if __has_builtin(__builtin_amdgcn_cvt_pk_bf16_f32)
  v2bf t = __builtin_amdgcn_cvt_pk_bf16_f32(a, b);   // RNE, 1 instr
  return __builtin_bit_cast(unsigned, t);
#else
  union { float f; unsigned u; } x, y; x.f = a; y.f = b;
  return __builtin_amdgcn_perm(y.u, x.u, 0x07060302u); // 1x v_perm_b32 (truncate)
#endif
}

__global__ __launch_bounds__(256) void mlp_semantic_kernel(
    const float* __restrict__ features,   // (B*N*K, F_IN)
    const float* __restrict__ spatial,    // (B*N*K, S_DIM)
    const float* __restrict__ W1,         // (S_DIM, F_IN+S_DIM)
    const float* __restrict__ b1,         // (S_DIM)
    const float* __restrict__ W2,         // (F_OUT, S_DIM)
    const float* __restrict__ b2,         // (F_OUT)
    float* __restrict__ out)              // (B*N, F_OUT)
{
  __shared__ unsigned sW1[S_DIM * W1_STRIDE];     // bf16 W1[s][f] pairs
  __shared__ unsigned sW2[F_OUT * W2_STRIDE];     // bf16 W2[o][s] pairs
  __shared__ float    sB1[S_DIM];
  __shared__ float    sB2[F_OUT];
  __shared__ float    sHsum[BN_PER_BLOCK * HS_STRIDE]; // fp32 hsum[bnLocal][s]

  const int tid = threadIdx.x;

  // ---- stage weights fp32 -> packed bf16 in LDS ----
  for (int i = tid; i < S_DIM * 96; i += 256) {
    int s = i / 96, p = i % 96;
    float2 w = ((const float2*)(W1 + s * (F_IN + S_DIM)))[p];
    sW1[s * W1_STRIDE + p] = pk_bf16(w.x, w.y);
  }
  for (int i = tid; i < F_OUT * 32; i += 256) {
    int o = i / 32, p = i % 32;
    float2 w = ((const float2*)(W2 + o * S_DIM))[p];
    sW2[o * W2_STRIDE + p] = pk_bf16(w.x, w.y);
  }
  if (tid < S_DIM) sB1[tid] = b1[tid];
  sB2[tid] = b2[tid];
  __syncthreads();

  const int lane = tid & 31;
  const int wave = tid >> 5;
  const int m    = lane & 15;
  const int half = lane >> 4;
  const long bnBase = (long)blockIdx.x * BN_PER_BLOCK;

  // ======== GEMM1: 2 bn pairs per wave, B-fragments loaded ONCE per slice ====
  const float* __restrict__ featRow[2];
  const float* __restrict__ spatRow[2];
  #pragma unroll
  for (int i = 0; i < 2; ++i) {
    const long bn = bnBase + 2 * wave + i;
    featRow[i] = features + (bn * KROWS + m) * F_IN;
    spatRow[i] = spatial  + (bn * KROWS + m) * S_DIM;
  }

  v8f acc[2][4];
  #pragma unroll
  for (int i = 0; i < 2; ++i)
    #pragma unroll
    for (int t = 0; t < 4; ++t)
      #pragma unroll
      for (int v = 0; v < 8; ++v) acc[i][t][v] = 0.0f;

  #pragma unroll
  for (int kk = 0; kk < 6; ++kk) {                // 6 x 32 over 192
    // 4 W1 B-fragments for this k-slice, shared by both bn pairs
    Frag bfr[4];
    #pragma unroll
    for (int t = 0; t < 4; ++t) {
      int s = t * 16 + m;                         // B frag: n = lane&15
      #pragma unroll
      for (int v = 0; v < 8; ++v)                 // k = kk*32 + half*16 + 2v
        bfr[t].u[v] = sW1[s * W1_STRIDE + kk * 16 + half * 8 + v];
    }
    #pragma unroll
    for (int i = 0; i < 2; ++i) {
      const float* src = (kk < 4) ? (featRow[i] + kk * 32)
                                  : (spatRow[i] + (kk - 4) * 32);
      // A frag: v -> k=16*half+2v (matches B)
      float4 q0 = *(const float4*)(src + half * 16);
      float4 q1 = *(const float4*)(src + half * 16 + 4);
      float4 q2 = *(const float4*)(src + half * 16 + 8);
      float4 q3 = *(const float4*)(src + half * 16 + 12);
      Frag a;
      a.u[0] = pk_bf16(q0.x, q0.y); a.u[1] = pk_bf16(q0.z, q0.w);
      a.u[2] = pk_bf16(q1.x, q1.y); a.u[3] = pk_bf16(q1.z, q1.w);
      a.u[4] = pk_bf16(q2.x, q2.y); a.u[5] = pk_bf16(q2.z, q2.w);
      a.u[6] = pk_bf16(q3.x, q3.y); a.u[7] = pk_bf16(q3.z, q3.w);
      #pragma unroll
      for (int t = 0; t < 4; ++t)
        acc[i][t] = __builtin_amdgcn_wmma_f32_16x16x32_bf16(
            false, a.v, false, bfr[t].v, (short)0, acc[i][t], false, false);
    }
  }

  // bias + ReLU + row-reduce (sum over the 16 k-rows) -> hsum fp32 in LDS
  #pragma unroll
  for (int i = 0; i < 2; ++i) {
    const int bnLocal = 2 * wave + i;
    #pragma unroll
    for (int t = 0; t < 4; ++t) {
      int s = t * 16 + m;
      float bv = sB1[s];
      float partial = 0.0f;
      #pragma unroll
      for (int v = 0; v < 8; ++v) partial += fmaxf(acc[i][t][v] + bv, 0.0f);
      partial += __shfl_xor(partial, 16, 32);     // rows 0-7 + rows 8-15
      if (lane < 16) sHsum[bnLocal * HS_STRIDE + s] = partial;
    }
  }
  __syncthreads();

  // ======== GEMM2: out(16 x 256) = hsum(16 x 64) * W2^T ========
  Frag a2[2];
  #pragma unroll
  for (int kk = 0; kk < 2; ++kk)
    #pragma unroll
    for (int v = 0; v < 8; ++v) {
      int sidx = kk * 32 + half * 16 + v * 2;
      float2 h2 = *(const float2*)(&sHsum[m * HS_STRIDE + sidx]);
      a2[kk].u[v] = pk_bf16(h2.x, h2.y);
    }

  #pragma unroll
  for (int j = 0; j < 2; ++j) {                   // 2 o-tiles per wave
    const int ot = wave + j * WAVES;
    v8f accO;
    #pragma unroll
    for (int v = 0; v < 8; ++v) accO[v] = 0.0f;
    #pragma unroll
    for (int kk = 0; kk < 2; ++kk) {
      Frag bfr;
      int o = ot * 16 + m;
      #pragma unroll
      for (int v = 0; v < 8; ++v)                 // s = kk*32 + half*16 + 2v
        bfr.u[v] = sW2[o * W2_STRIDE + kk * 16 + half * 8 + v];
      accO = __builtin_amdgcn_wmma_f32_16x16x32_bf16(
          false, a2[kk].v, false, bfr.v, (short)0, accO, false, false);
    }
    const int o = ot * 16 + m;
    const float bb = 16.0f * sB2[o];              // b2 summed K=16 times
    #pragma unroll
    for (int v = 0; v < 8; ++v) {                 // C row = bnLocal
      int r = v + half * 8;
      out[(bnBase + r) * (long)F_OUT + o] = accO[v] + bb;
    }
  }
}

extern "C" void kernel_launch(void* const* d_in, const int* in_sizes, int n_in,
                              void* d_out, int out_size, void* d_ws, size_t ws_size,
                              hipStream_t stream) {
  const float* features = (const float*)d_in[0];
  const float* spatial  = (const float*)d_in[1];
  const float* W1       = (const float*)d_in[2];
  const float* b1       = (const float*)d_in[3];
  const float* W2       = (const float*)d_in[4];
  const float* b2       = (const float*)d_in[5];
  float* out            = (float*)d_out;

  const int BN = 8 * 4096;
  dim3 grid(BN / BN_PER_BLOCK), block(256);
  mlp_semantic_kernel<<<grid, block, 0, stream>>>(
      features, spatial, W1, b1, W2, b2, out);
}
